// GatEncoder_47880295416164
// MI455X (gfx1250) — compile-verified, matrix operand order checked by reading, not run on hardware
//
#include <hip/hip_runtime.h>

typedef __attribute__((ext_vector_type(16))) _Float16 v16h;
typedef __attribute__((ext_vector_type(8)))  _Float16 v8h;
typedef __attribute__((ext_vector_type(8)))  float    v8f;

#define HEADS 4
#define CH    64
#define GHC   256

__device__ __forceinline__ float leaky02(float x) { return x > 0.f ? x : 0.2f * x; }

// float atomic max via sign-split int/uint ordering trick (hardware atomics)
__device__ __forceinline__ void atomicMaxFloat(float* addr, float val) {
  if (val >= 0.f) atomicMax((int*)addr, __float_as_int(val));
  else            atomicMin((unsigned int*)addr, __float_as_uint(val));
}

// ---------------- fill ----------------
__global__ void k_fill(float* __restrict__ p, float v, long n) {
  long i = (long)blockIdx.x * blockDim.x + threadIdx.x;
  if (i < n) p[i] = v;
}

// ---------------- f32 -> f16 convert, optional periodic bias add ----------------
__global__ void k_cvt16(const float* __restrict__ src, _Float16* __restrict__ dst,
                        const float* __restrict__ bias, int width, long n) {
  long i = (long)blockIdx.x * blockDim.x + threadIdx.x;
  if (i < n) {
    float v = src[i];
    if (bias) v += bias[i % width];
    dst[i] = (_Float16)v;
  }
}

// ---------------- transpose-convert weights: W[K x N] f32 -> Wt[N x K] f16 ----------------
__global__ void k_cvt16_T(const float* __restrict__ src, _Float16* __restrict__ dst,
                          int K, int N) {
  int i = blockIdx.x * blockDim.x + threadIdx.x;
  if (i < K * N) {
    int n = i / K, k = i - n * K;
    dst[i] = (_Float16)src[(long)k * N + n];
  }
}

// ---------------- WMMA GEMM: C[M x N] = A[M x K] * Bt[N x K]^T (+bias, +relu) ----------------
// One wave per 16x16 output tile. M % 16 == 0, N % 16 == 0, K % 32 == 0.
// f16 A/B fragment lane layout (ISA 7.12.2): lane L holds row/col (L&15);
// vector elems 0..7 -> K = kb+0..7, elems 8..15 -> K = kb+16..23, kb = 8*(L>>4).
__global__ void k_wmma_gemm(const _Float16* __restrict__ A,
                            const _Float16* __restrict__ Bt,
                            const float* __restrict__ bias,
                            float* __restrict__ C,
                            int M, int N, int K, int relu) {
  const int gid    = blockIdx.x * blockDim.x + threadIdx.x;
  const int wave   = gid >> 5;
  const int lane   = threadIdx.x & 31;
  const int ntiles = N >> 4;
  const int tm = wave / ntiles;
  const int tn = wave - tm * ntiles;
  if (tm * 16 >= M) return;   // wave-uniform

  const int r  = lane & 15;
  const int kb = (lane >> 4) << 3;
  const _Float16* arow = A  + (size_t)(tm * 16 + r) * K + kb;
  const _Float16* brow = Bt + (size_t)(tn * 16 + r) * K + kb;

  v8f acc = {};
  for (int k0 = 0; k0 < K; k0 += 32) {
    v8h alo = *(const v8h*)(arow + k0);
    v8h ahi = *(const v8h*)(arow + k0 + 16);
    v8h blo = *(const v8h*)(brow + k0);
    v8h bhi = *(const v8h*)(brow + k0 + 16);
    v16h a, b;
#pragma unroll
    for (int i = 0; i < 8; ++i) {
      a[i] = alo[i]; a[i + 8] = ahi[i];
      b[i] = blo[i]; b[i + 8] = bhi[i];
    }
    acc = __builtin_amdgcn_wmma_f32_16x16x32_f16(false, a, false, b,
                                                 (short)0, acc, false, false);
  }

  const int col   = tn * 16 + r;
  const float bv  = bias ? bias[col] : 0.f;
  const int rbase = tm * 16 + ((lane >> 4) << 3);  // C/D: reg v -> row v + 8*(L>>4)
#pragma unroll
  for (int v = 0; v < 8; ++v) {
    float o = acc[v] + bv;
    if (relu) o = fmaxf(o, 0.f);
    C[(size_t)(rbase + v) * N + col] = o;
  }
}

// ---------------- per-(node,head) attention logits s, d ----------------
__global__ void k_node_logits(const float* __restrict__ h, const float* __restrict__ as,
                              const float* __restrict__ ad, float* __restrict__ s,
                              float* __restrict__ d, int N) {
  int i = blockIdx.x * blockDim.x + threadIdx.x;
  if (i >= N * HEADS) return;
  int head = i & 3;
  const float* hp = h + (size_t)(i >> 2) * GHC + head * CH;
  const float* ap = as + head * CH;
  const float* bp = ad + head * CH;
  float ss = 0.f, dd = 0.f;
#pragma unroll 8
  for (int c = 0; c < CH; ++c) { float v = hp[c]; ss += v * ap[c]; dd += v * bp[c]; }
  s[i] = ss; d[i] = dd;
}

// ---------------- per-(edge,head): segment max of leaky-relu logits ----------------
__global__ void k_edge_max(const int* __restrict__ ei, int E, int N,
                           const float* __restrict__ s, const float* __restrict__ d,
                           float* __restrict__ m) {
  int t = blockIdx.x * blockDim.x + threadIdx.x;
  if (t >= (E + N) * HEADS) return;
  int e = t >> 2, head = t & 3;
  int sv = e < E ? ei[e]     : e - E;   // self-loops appended
  int dv = e < E ? ei[E + e] : e - E;
  float logit = leaky02(s[sv * 4 + head] + d[dv * 4 + head]);
  atomicMaxFloat(m + dv * 4 + head, logit);
}

// ---------------- per-(edge,head): exp(logit - max), segment-sum denom ----------------
__global__ void k_edge_exp(const int* __restrict__ ei, int E, int N,
                           const float* __restrict__ s, const float* __restrict__ d,
                           const float* __restrict__ m, float* __restrict__ den,
                           float* __restrict__ ex) {
  int t = blockIdx.x * blockDim.x + threadIdx.x;
  if (t >= (E + N) * HEADS) return;
  int e = t >> 2, head = t & 3;
  int sv = e < E ? ei[e]     : e - E;
  int dv = e < E ? ei[E + e] : e - E;
  float logit = leaky02(s[sv * 4 + head] + d[dv * 4 + head]);
  float v = expf(logit - m[dv * 4 + head]);
  ex[t] = v;
  atomicAdd(den + dv * 4 + head, v);
}

// ---------------- one wave32 per edge: out[dst] += h[src] * alpha (8 ch / lane) ----------------
__global__ void k_edge_agg(const int* __restrict__ ei, int E, int N,
                           const float* __restrict__ h, const float* __restrict__ ex,
                           const float* __restrict__ den, float* __restrict__ out) {
  int gid  = blockIdx.x * blockDim.x + threadIdx.x;
  int e    = gid >> 5;
  int lane = threadIdx.x & 31;
  if (e >= E + N) return;
  int sv = e < E ? ei[e]     : e - E;
  int dv = e < E ? ei[E + e] : e - E;
  int head = lane >> 3;                 // 8 lanes per head, 8 channels per lane
  float alpha = ex[(size_t)e * 4 + head] / fmaxf(den[dv * 4 + head], 1e-16f);
  const float4* hp = (const float4*)(h + (size_t)sv * GHC + lane * 8);
  float4 v0 = hp[0], v1 = hp[1];
  float* op = out + (size_t)dv * GHC + lane * 8;
  atomicAdd(op + 0, v0.x * alpha); atomicAdd(op + 1, v0.y * alpha);
  atomicAdd(op + 2, v0.z * alpha); atomicAdd(op + 3, v0.w * alpha);
  atomicAdd(op + 4, v1.x * alpha); atomicAdd(op + 5, v1.y * alpha);
  atomicAdd(op + 6, v1.z * alpha); atomicAdd(op + 7, v1.w * alpha);
}

// ---------------- global mean pool (segment sum + count) ----------------
__global__ void k_pool(const float* __restrict__ y, const int* __restrict__ batch,
                       float* __restrict__ pool, float* __restrict__ cnt, int N) {
  int n = blockIdx.x;
  if (n >= N) return;
  int g = batch[n];
  atomicAdd(pool + (size_t)g * CH + threadIdx.x, y[(size_t)n * CH + threadIdx.x]);
  if (threadIdx.x == 0) atomicAdd(cnt + g, 1.f);
}

__global__ void k_pool_div(float* __restrict__ pool, const float* __restrict__ cnt, int G) {
  int i = blockIdx.x * blockDim.x + threadIdx.x;
  if (i < G * CH) pool[i] /= fmaxf(cnt[i >> 6], 1.f);  // CH == 64
}

// ================================================================
extern "C" void kernel_launch(void* const* d_in, const int* in_sizes, int n_in,
                              void* d_out, int out_size, void* d_ws, size_t ws_size,
                              hipStream_t stream) {
  const float* x     = (const float*)d_in[0];
  const int*   ei    = (const int*)d_in[1];
  const int*   batch = (const int*)d_in[2];
  const int N  = in_sizes[0] / 32;
  const int E  = in_sizes[1] / 2;
  const int G  = out_size / 256;
  const int ET = E + N;

  const float* W[3]   = {(const float*)d_in[4],  (const float*)d_in[10], (const float*)d_in[16]};
  const float* As[3]  = {(const float*)d_in[5],  (const float*)d_in[11], (const float*)d_in[17]};
  const float* Ad[3]  = {(const float*)d_in[6],  (const float*)d_in[12], (const float*)d_in[18]};
  const float* Bb[3]  = {(const float*)d_in[7],  (const float*)d_in[13], (const float*)d_in[19]};
  const float* HLW[3] = {(const float*)d_in[8],  (const float*)d_in[14], (const float*)d_in[20]};
  const float* HLB[3] = {(const float*)d_in[9],  (const float*)d_in[15], (const float*)d_in[21]};
  const float* mh1w = (const float*)d_in[22];
  const float* mh1b = (const float*)d_in[23];
  const float* mh2w = (const float*)d_in[24];
  const float* mh2b = (const float*)d_in[25];

  // workspace carve-up (256B aligned)
  char* wp = (char*)d_ws;
  auto carve = [&](size_t bytes) -> char* {
    char* p = wp; wp += (bytes + 255) & ~(size_t)255; return p;
  };
  float*    h    = (float*)   carve((size_t)N * GHC * 4);
  float*    h2   = (float*)   carve((size_t)N * GHC * 4);
  _Float16* a16  = (_Float16*)carve((size_t)N * GHC * 2);
  _Float16* w16  = (_Float16*)carve((size_t)GHC * GHC * 2);
  float*    sl   = (float*)   carve((size_t)N * 4 * 4);
  float*    dl   = (float*)   carve((size_t)N * 4 * 4);
  float*    mx   = (float*)   carve((size_t)N * 4 * 4);
  float*    den  = (float*)   carve((size_t)ET * 0 + (size_t)N * 4 * 4);
  float*    exv  = (float*)   carve((size_t)ET * 4 * 4);
  float*    y    = (float*)   carve((size_t)N * CH * 4);
  float*    pool = (float*)   carve((size_t)G * CH * 4);
  float*    cnt  = (float*)   carve((size_t)G * 4);
  float*    t1   = (float*)   carve((size_t)G * GHC * 4);
  (void)ws_size; (void)n_in;

  auto nb = [](long n, int b) { return dim3((unsigned)((n + b - 1) / b)); };
  auto gemm = [&](const _Float16* A, const _Float16* Bt, const float* bias, float* C,
                  int M, int Nn, int K, int relu) {
    long thr = (long)(M / 16) * (Nn / 16) * 32;
    k_wmma_gemm<<<nb(thr, 256), 256, 0, stream>>>(A, Bt, bias, C, M, Nn, K, relu);
  };

  const float NEG_INF = -__builtin_huge_valf();
  const float* cur = x;
  int K = 32;
  for (int l = 0; l < 3; ++l) {
    // h = cur @ W_l   (f16 WMMA, f32 accum)
    k_cvt16<<<nb((long)N * K, 256), 256, 0, stream>>>(cur, a16, nullptr, 1, (long)N * K);
    k_cvt16_T<<<nb((long)K * GHC, 256), 256, 0, stream>>>(W[l], w16, K, GHC);
    gemm(a16, w16, nullptr, h, N, GHC, K, 0);
    // attention logits
    k_node_logits<<<nb((long)N * 4, 256), 256, 0, stream>>>(h, As[l], Ad[l], sl, dl, N);
    // segment softmax + aggregate
    k_fill<<<nb((long)N * 4, 256), 256, 0, stream>>>(mx, NEG_INF, (long)N * 4);
    k_fill<<<nb((long)N * 4, 256), 256, 0, stream>>>(den, 0.f, (long)N * 4);
    k_fill<<<nb((long)N * GHC, 256), 256, 0, stream>>>(h2, 0.f, (long)N * GHC);
    k_edge_max<<<nb((long)ET * 4, 256), 256, 0, stream>>>(ei, E, N, sl, dl, mx);
    k_edge_exp<<<nb((long)ET * 4, 256), 256, 0, stream>>>(ei, E, N, sl, dl, mx, den, exv);
    k_edge_agg<<<nb((long)ET * 32, 256), 256, 0, stream>>>(ei, E, N, h, exv, den, h2);
    // y = relu((h2 + b_l) @ hl_w + hl_b)   (no relu after layer 3)
    k_cvt16<<<nb((long)N * GHC, 256), 256, 0, stream>>>(h2, a16, Bb[l], GHC, (long)N * GHC);
    k_cvt16_T<<<nb((long)GHC * CH, 256), 256, 0, stream>>>(HLW[l], w16, GHC, CH);
    gemm(a16, w16, HLB[l], y, N, CH, GHC, l < 2 ? 1 : 0);
    cur = y; K = CH;
  }

  // global mean pool
  k_fill<<<nb((long)G * CH, 256), 256, 0, stream>>>(pool, 0.f, (long)G * CH);
  k_fill<<<nb((long)G, 256), 256, 0, stream>>>(cnt, 0.f, (long)G);
  k_pool<<<dim3((unsigned)N), dim3(CH), 0, stream>>>(y, batch, pool, cnt, N);
  k_pool_div<<<nb((long)G * CH, 256), 256, 0, stream>>>(pool, cnt, G);

  // head MLP
  k_cvt16<<<nb((long)G * CH, 256), 256, 0, stream>>>(pool, a16, nullptr, 1, (long)G * CH);
  k_cvt16_T<<<nb((long)CH * 256, 256), 256, 0, stream>>>(mh1w, w16, CH, 256);
  gemm(a16, w16, mh1b, t1, G, 256, CH, 1);
  k_cvt16<<<nb((long)G * 256, 256), 256, 0, stream>>>(t1, a16, nullptr, 1, (long)G * 256);
  k_cvt16_T<<<nb((long)256 * 256, 256), 256, 0, stream>>>(mh2w, w16, 256, 256);
  gemm(a16, w16, mh2b, (float*)d_out, G, 256, 256, 0);
}
